// Model_39676907883015
// MI455X (gfx1250) — compile-verified, matrix operand order checked by reading, not run on hardware
//
#include <hip/hip_runtime.h>
#include <hip/hip_bf16.h>

typedef __attribute__((ext_vector_type(16))) __bf16 v16bf;
typedef __attribute__((ext_vector_type(8)))  float  v8f;
typedef __attribute__((ext_vector_type(8)))  unsigned int v8u;

#define SDIM 2048
#define DDIM 64
#define QTILE_WAVE 16
#define WAVES 8
#define QTILE_BLOCK 128   // WAVES * QTILE_WAVE
#define KTILE 64          // keys per mainloop iteration
#define NT (SDIM / KTILE) // 32 mainloop iterations
#define KT_STRIDE 66      // K tile: 64 + 2 pad (ushorts)
#define VTP_STRIDE 65     // V tile: key-pair packed rows, 64 + 1 pad (uints)
#define P_STRIDE 66       // P scratch: 64 + 2 pad (ushorts)

// one-instruction bf16x2 pack: v_perm_b32 selecting high halves of two f32
__device__ __forceinline__ unsigned pack2bf(float lo, float hi) {
  return __builtin_amdgcn_perm(__builtin_bit_cast(unsigned, hi),
                               __builtin_bit_cast(unsigned, lo),
                               0x07060302u);
}
__device__ __forceinline__ unsigned short f2bf(float f) {
  return (unsigned short)(__builtin_bit_cast(unsigned, f) >> 16);
}
__device__ __forceinline__ float xor_bcast(float v, int mask) {
  int idx = (((int)(threadIdx.x & 31u) ^ mask) << 2);
  return __builtin_bit_cast(float,
      __builtin_amdgcn_ds_bpermute(idx, __builtin_bit_cast(int, v)));
}

__global__ __launch_bounds__(256)
void fa_fwd_bf16wmma(const float* __restrict__ Q, const float* __restrict__ K,
                     const float* __restrict__ V, float* __restrict__ O) {
  __shared__ unsigned short kt[KTILE * KT_STRIDE];
  __shared__ unsigned       vtp[(KTILE / 2) * VTP_STRIDE];  // key-pair packed
  __shared__ unsigned short ps[WAVES][16 * P_STRIDE];

  const int tid  = threadIdx.x;
  const int lane = tid & 31;
  const int wave = tid >> 5;
  const int n    = lane & 15;   // N / M index inside 16-lane half
  const int hh   = lane >> 4;   // half-wave select

  const int blocks_per_bh = SDIM / QTILE_BLOCK;       // 16
  const int bh   = blockIdx.x / blocks_per_bh;        // fused batch*head (64)
  const int qblk = blockIdx.x % blocks_per_bh;
  const size_t base = (size_t)bh * SDIM * DDIM;
  const int q0 = qblk * QTILE_BLOCK + wave * QTILE_WAVE;

  const float* Kb = K + base;
  const float* Vb = V + base;

  // fold softmax scale (1/sqrt(64)) and log2(e) into Q so we can use exp2
  const float QSCALE = 0.125f * 1.44269504088896340736f;

  // ---- Q fragments: A-layout 16x32 bf16, two K-chunks covering D=64 ----
  v8u qa_u[2];
  {
    const float* qrow = Q + base + (size_t)(q0 + n) * DDIM;   // row M = n
    #pragma unroll
    for (int c = 0; c < 2; ++c) {
      #pragma unroll
      for (int v = 0; v < 8; ++v) {
        int k = (c << 5) + ((v >> 2) << 4) + (hh << 3) + ((v & 3) << 1);
        qa_u[c][v] = pack2bf(qrow[k] * QSCALE, qrow[k + 1] * QSCALE);
      }
    }
  }
  const v16bf qa0 = __builtin_bit_cast(v16bf, qa_u[0]);
  const v16bf qa1 = __builtin_bit_cast(v16bf, qa_u[1]);

  // all-ones B fragment: rowsum(P) = P @ ones via the XDL pipe
  v8u ones_u;
  #pragma unroll
  for (int i = 0; i < 8; ++i) ones_u[i] = 0x3F803F80u;   // bf16 1.0 x2
  const v16bf onesb = __builtin_bit_cast(v16bf, ones_u);

  float m_i[8];
  v8f o_acc[4], l_acc;
  const v8f vzero = {};
  #pragma unroll
  for (int r = 0; r < 8; ++r) m_i[r] = -__builtin_inff();
  #pragma unroll
  for (int nd = 0; nd < 4; ++nd) o_acc[nd] = vzero;
  l_acc = vzero;

  // ---- register double-buffer for the K/V tile global loads ----
  float2 kreg[8];
  float  vreg[8][2];
  auto load_tile = [&](int key0) {
    const float* kb = Kb + (size_t)key0 * DDIM;
    const float* vb = Vb + (size_t)key0 * DDIM;
    #pragma unroll
    for (int i = 0; i < 8; ++i) {
      int p = tid + (i << 8);           // 0..2047 float2-pairs
      int row = p >> 5;                 // key in tile (0..63)
      int d   = (p & 31) << 1;          // even dim
      kreg[i] = *(const float2*)(kb + row * DDIM + d);
    }
    #pragma unroll
    for (int i = 0; i < 8; ++i) {
      int p  = tid + (i << 8);          // 0..2047 (keypair, d) cells
      int rp = p >> 6;                  // key pair (0..31)
      int d  = p & 63;                  // dim
      vreg[i][0] = vb[(rp * 2)     * DDIM + d];
      vreg[i][1] = vb[(rp * 2 + 1) * DDIM + d];
    }
  };
  auto store_tile = [&]() {
    #pragma unroll
    for (int i = 0; i < 8; ++i) {
      int p = tid + (i << 8);
      int row = p >> 5;
      int d   = (p & 31) << 1;
      *(unsigned*)&kt[row * KT_STRIDE + d] = pack2bf(kreg[i].x, kreg[i].y);
    }
    #pragma unroll
    for (int i = 0; i < 8; ++i) {
      int p  = tid + (i << 8);
      int rp = p >> 6;
      int d  = p & 63;
      vtp[rp * VTP_STRIDE + d] = pack2bf(vreg[i][0], vreg[i][1]);
    }
  };

  load_tile(0);                         // prologue: tile 0 in flight

  for (int t = 0; t < NT; ++t) {
    __syncthreads();                    // previous tile's readers retired
    store_tile();                       // stage tile t (regs -> LDS bf16)
    __syncthreads();                    // tile t visible to all waves
    if (t + 1 < NT) load_tile((t + 1) * KTILE);   // hide HBM behind compute
    if (t + 2 < NT)                               // keep L2 one tile ahead
      __builtin_prefetch(Kb + (size_t)(t + 2) * KTILE * DDIM + tid * 16, 0, 1);

    // ---- S = Q K^T : four 16x16 score tiles over 64 keys ----
    v8f s_acc[4];
    #pragma unroll
    for (int nt = 0; nt < 4; ++nt) {
      v8u kb_u[2];
      const int key = (nt << 4) + n;    // column N = key within tile
      #pragma unroll
      for (int c = 0; c < 2; ++c) {
        #pragma unroll
        for (int v = 0; v < 8; ++v) {
          int kk = ((v >> 2) << 4) + (hh << 3) + ((v & 3) << 1);
          kb_u[c][v] = *(const unsigned*)&kt[key * KT_STRIDE + (c << 5) + kk];
        }
      }
      v8f acc = vzero;
      acc = __builtin_amdgcn_wmma_f32_16x16x32_bf16(false, qa0, false,
              __builtin_bit_cast(v16bf, kb_u[0]), (short)0, acc, false, false);
      acc = __builtin_amdgcn_wmma_f32_16x16x32_bf16(false, qa1, false,
              __builtin_bit_cast(v16bf, kb_u[1]), (short)0, acc, false, false);
      s_acc[nt] = acc;
    }

    // ---- online softmax (log2 domain), rows M = r + 8*hh ----
    // only the row-MAX needs a cross-lane tree; row-SUM goes through WMMA
    float alpha[8];
    #pragma unroll
    for (int r = 0; r < 8; ++r) {
      float s0 = s_acc[0][r], s1 = s_acc[1][r];
      float s2 = s_acc[2][r], s3 = s_acc[3][r];
      float mx = fmaxf(fmaxf(s0, s1), fmaxf(s2, s3));
      mx = fmaxf(mx, xor_bcast(mx, 1));
      mx = fmaxf(mx, xor_bcast(mx, 2));
      mx = fmaxf(mx, xor_bcast(mx, 4));
      mx = fmaxf(mx, xor_bcast(mx, 8));
      float mn = fmaxf(m_i[r], mx);
      float a = exp2f(m_i[r] - mn);     // = 0 on first tile (m_i == -inf)
      m_i[r] = mn;
      alpha[r] = a;
      int row = (r + (hh << 3)) * P_STRIDE;
      ps[wave][row + n]      = f2bf(exp2f(s0 - mn));
      ps[wave][row + 16 + n] = f2bf(exp2f(s1 - mn));
      ps[wave][row + 32 + n] = f2bf(exp2f(s2 - mn));
      ps[wave][row + 48 + n] = f2bf(exp2f(s3 - mn));
    }
    #pragma unroll
    for (int nd = 0; nd < 4; ++nd)
      #pragma unroll
      for (int r = 0; r < 8; ++r)
        o_acc[nd][r] *= alpha[r];
    #pragma unroll
    for (int r = 0; r < 8; ++r)
      l_acc[r] *= alpha[r];

    // per-wave LDS round trip: C-layout P -> A-layout bf16 fragments
    asm volatile("s_wait_dscnt 0" ::: "memory");

    v16bf pa[2];
    #pragma unroll
    for (int c = 0; c < 2; ++c) {
      v8u pa_u;
      #pragma unroll
      for (int v = 0; v < 8; ++v) {
        int k = (c << 5) + ((v >> 2) << 4) + (hh << 3) + ((v & 3) << 1);
        pa_u[v] = *(const unsigned*)&ps[wave][n * P_STRIDE + k];
      }
      pa[c] = __builtin_bit_cast(v16bf, pa_u);
    }

    // ---- O += P V : four 16x16 d-tiles, K = 64 keys (2 chunks) ----
    #pragma unroll
    for (int nd = 0; nd < 4; ++nd) {
      #pragma unroll
      for (int c = 0; c < 2; ++c) {
        v8u vb_u;
        #pragma unroll
        for (int v = 0; v < 8; ++v) {
          // key pair index = kk/2, kk = 32c + 16*(v>>2) + 8*hh + 2*(v&3)
          int kkp = (c << 4) + ((v >> 2) << 3) + (hh << 2) + (v & 3);
          vb_u[v] = vtp[kkp * VTP_STRIDE + (nd << 4) + n];
        }
        o_acc[nd] = __builtin_amdgcn_wmma_f32_16x16x32_bf16(false, pa[c], false,
                      __builtin_bit_cast(v16bf, vb_u), (short)0, o_acc[nd],
                      false, false);
      }
    }
    // ---- l += P @ ones : row sums on the XDL pipe ----
    l_acc = __builtin_amdgcn_wmma_f32_16x16x32_bf16(false, pa[0], false,
              onesb, (short)0, l_acc, false, false);
    l_acc = __builtin_amdgcn_wmma_f32_16x16x32_bf16(false, pa[1], false,
              onesb, (short)0, l_acc, false, false);
  }

  // ---- epilogue: O / l ----
  float* orow = O + base + (size_t)q0 * DDIM;
  #pragma unroll
  for (int r = 0; r < 8; ++r) {
    float inv = 1.0f / l_acc[r];
    int row = (r + (hh << 3)) * DDIM;
    #pragma unroll
    for (int nd = 0; nd < 4; ++nd)
      orow[row + (nd << 4) + n] = o_acc[nd][r] * inv;
  }
}

extern "C" void kernel_launch(void* const* d_in, const int* in_sizes, int n_in,
                              void* d_out, int out_size, void* d_ws, size_t ws_size,
                              hipStream_t stream) {
  (void)in_sizes; (void)n_in; (void)out_size; (void)d_ws; (void)ws_size;
  const float* q = (const float*)d_in[0];
  const float* k = (const float*)d_in[1];
  const float* v = (const float*)d_in[2];
  float* o = (float*)d_out;
  // grid = B*H * (S / QTILE_BLOCK) = 64 * 16 = 1024 blocks of 256 threads
  dim3 grid(64 * (SDIM / QTILE_BLOCK));
  dim3 block(256);
  fa_fwd_bf16wmma<<<grid, block, 0, stream>>>(q, k, v, o);
}
